// Attention_69818988363776
// MI455X (gfx1250) — matrix operand order repaired and checked by reading, compiled, not run on hardware
//
#include <hip/hip_runtime.h>

typedef __attribute__((ext_vector_type(16))) __bf16 v16bf;
typedef __attribute__((ext_vector_type(8)))  float  v8f;
typedef __attribute__((ext_vector_type(4)))  int    v4i;

#define S_LEN  2048
#define D_DIM  128
#define QTILE  16
#define KTILE  32
#define WAVES  8
#define SCALE  0.08838834764831845f   // 1/sqrt(128)
#define NEGBIG (-3.0e38f)

union BF16Frag {
    unsigned int u[8];
    v16bf        v;
};

static __device__ __forceinline__ unsigned int fbits(float f) {
    return __builtin_bit_cast(unsigned int, f);
}
// pack two floats to packed bf16 (round-half-up) with one v_perm_b32
static __device__ __forceinline__ unsigned int pack_bf16(float lo, float hi) {
    return __builtin_amdgcn_perm(fbits(hi) + 0x8000u, fbits(lo) + 0x8000u, 0x07060302u);
}
static __device__ __forceinline__ unsigned short f2bf(float f) {
    return (unsigned short)((fbits(f) + 0x8000u) >> 16);
}

__global__ __launch_bounds__(256) void
fattn_wmma_bf16(const float* __restrict__ qg,
                const float* __restrict__ kg,
                const float* __restrict__ vg,
                const unsigned char* __restrict__ maskg,
                float* __restrict__ outg)
{
    // double-buffered cooperative tiles
    __shared__ unsigned int klds[2][KTILE][D_DIM / 2 + 1];             // ~16.6 KB
    __shared__ unsigned int vlds[2][D_DIM][KTILE / 2 + 1];             // ~17.4 KB
    __shared__ __align__(16) unsigned char mlds[2][QTILE * WAVES][KTILE]; // 8 KB
    // per-wave P staging (16 x 32 bf16, padded)
    __shared__ unsigned short pstage[WAVES][QTILE][KTILE + 2];         // ~8.7 KB

    const int tid  = threadIdx.x;
    const int lane = tid & 31;
    const int wave = tid >> 5;
    const int hh   = lane >> 4;      // lane half (0/1)
    const int lm   = lane & 15;

    const int nQB    = S_LEN / (QTILE * WAVES);        // 16 q-blocks per batch
    const int b      = blockIdx.x / nQB;
    const int qBlock = blockIdx.x % nQB;
    const int qBase  = qBlock * (QTILE * WAVES) + wave * QTILE;

    const float* qb  = qg + ((size_t)b * S_LEN + qBase) * D_DIM;
    const float* kb  = kg + (size_t)b * S_LEN * D_DIM;
    const float* vb  = vg + (size_t)b * S_LEN * D_DIM;
    const unsigned char* mbB =
        maskg + ((size_t)b * S_LEN + (size_t)qBlock * (QTILE * WAVES)) * (size_t)S_LEN;

    const int mr = tid >> 1;          // mask q-row within block (0..127)
    const int mh = tid & 1;           // 16-byte half of the 32-key mask row

    // stage tile starting at key 'tt' into buffer 'sb'
    auto stage = [&](int tt, int sb) {
        // async mask tile -> LDS (one b128 per thread)
        {
            const unsigned char* src = mbB + (size_t)mr * S_LEN + tt + 16 * mh;
            unsigned char*       dst = &mlds[sb][mr][16 * mh];
#if __has_builtin(__builtin_amdgcn_global_load_async_to_lds_b128)
            __builtin_amdgcn_global_load_async_to_lds_b128((v4i*)src, (v4i*)dst, 0, 0);
#else
            *(uint4*)dst = *(const uint4*)src;
#endif
        }
        // K tile as bf16 dim-pairs: klds[key][dp]
#pragma unroll
        for (int i = 0; i < 8; ++i) {
            const int p   = i * 256 + tid;
            const int key = p >> 6;
            const int dp  = p & 63;
            float2 f = *(const float2*)(kb + (size_t)(tt + key) * D_DIM + 2 * dp);
            klds[sb][key][dp] = pack_bf16(f.x, f.y);
        }
        // V tile transposed as bf16 key-pairs: vlds[dim][kp]
#pragma unroll
        for (int i = 0; i < 8; ++i) {
            const int u   = i * 256 + tid;
            const int kp  = u >> 7;
            const int dim = u & 127;
            const float x = vb[(size_t)(tt + 2 * kp)     * D_DIM + dim];
            const float y = vb[(size_t)(tt + 2 * kp + 1) * D_DIM + dim];
            vlds[sb][dim][kp] = pack_bf16(x, y);
        }
    };

    // ---- load Q once as 4 bf16 A-fragments (16x32 each, D=128) ----
    BF16Frag qf[4];
#pragma unroll
    for (int c = 0; c < 4; ++c) {
        const float* row = qb + (size_t)lm * D_DIM + c * 32;
#pragma unroll
        for (int vv = 0; vv < 8; ++vv) {
            const int k0 = 2 * vv + 16 * hh;
            float2 f = *(const float2*)(row + k0);
            qf[c].u[vv] = pack_bf16(f.x, f.y);
        }
    }

    // B-fragment that is 1.0 in column 0, 0 elsewhere (row-sum accumulator trick)
    BF16Frag ones_b;
    {
        const unsigned int oc = (lm == 0) ? 0x3F803F80u : 0u;   // bf16(1.0) pair
#pragma unroll
        for (int vv = 0; vv < 8; ++vv) ones_b.u[vv] = oc;
    }

    // online-softmax state: lane covers rows (j + 8*hh), j = 0..7
    float m_i[8];
    v8f   o_acc[8];      // 8 output dim-chunks of 16
    v8f   o_sum;         // running softmax denominator via ones-column WMMA
#pragma unroll
    for (int j = 0; j < 8; ++j) m_i[j] = NEGBIG;
#pragma unroll
    for (int c = 0; c < 8; ++c) o_acc[c] = (v8f){0,0,0,0,0,0,0,0};
    o_sum = (v8f){0,0,0,0,0,0,0,0};

    stage(0, 0);   // prologue: fill buffer 0

    int buf = 0;
    for (int t = 0; t < S_LEN; t += KTILE, buf ^= 1) {
#if __has_builtin(__builtin_amdgcn_global_load_async_to_lds_b128)
        asm volatile("s_wait_asynccnt 0" ::: "memory");
#endif
        __syncthreads();                       // staged tile 'buf' now visible

        if (t + KTILE < S_LEN) stage(t + KTILE, buf ^ 1);   // overlap next tile

        // ---- scores = Q * K^T over 32 keys (two 16x16 fp32 accumulators) ----
        v8f s0 = (v8f){0,0,0,0,0,0,0,0};
        v8f s1 = (v8f){0,0,0,0,0,0,0,0};
#pragma unroll
        for (int c = 0; c < 4; ++c) {
            BF16Frag kf0, kf1;
#pragma unroll
            for (int vv = 0; vv < 8; ++vv) {
                const int dp = c * 16 + vv + 8 * hh;
                kf0.u[vv] = klds[buf][lm][dp];
                kf1.u[vv] = klds[buf][16 + lm][dp];
            }
            s0 = __builtin_amdgcn_wmma_f32_16x16x32_bf16(false, qf[c].v, false, kf0.v,
                                                         (short)0, s0, false, false);
            s1 = __builtin_amdgcn_wmma_f32_16x16x32_bf16(false, qf[c].v, false, kf1.v,
                                                         (short)0, s1, false, false);
        }

        // ---- scale + mask + row-max ----
        float rmax[8];
#pragma unroll
        for (int j = 0; j < 8; ++j) {
            const int qr = j + 8 * hh;
            const unsigned char* mrow = &mlds[buf][wave * QTILE + qr][0];
            float a0 = mrow[lm]      ? NEGBIG : s0[j] * SCALE;
            float a1 = mrow[16 + lm] ? NEGBIG : s1[j] * SCALE;
            s0[j] = a0; s1[j] = a1;
            float mx = fmaxf(a0, a1);
            mx = fmaxf(mx, __shfl_xor(mx, 1, 32));
            mx = fmaxf(mx, __shfl_xor(mx, 2, 32));
            mx = fmaxf(mx, __shfl_xor(mx, 4, 32));
            mx = fmaxf(mx, __shfl_xor(mx, 8, 32));
            rmax[j] = mx;
        }

        // ---- online softmax update, P -> per-wave LDS (bf16) ----
        v8f corrv;
#pragma unroll
        for (int j = 0; j < 8; ++j) {
            const float mnew = fmaxf(m_i[j], rmax[j]);
            corrv[j] = __expf(m_i[j] - mnew);
            m_i[j]   = mnew;
            const float p0 = __expf(s0[j] - mnew);
            const float p1 = __expf(s1[j] - mnew);
            const int qr = j + 8 * hh;
            pstage[wave][qr][lm]      = f2bf(p0);
            pstage[wave][qr][16 + lm] = f2bf(p1);
        }
        // packed rescale of accumulators (v_pk_mul_f32 pairs)
#pragma unroll
        for (int c = 0; c < 8; ++c) o_acc[c] *= corrv;
        o_sum *= corrv;

        // per-wave LDS is in-order; fence so reload sees cross-lane stores
        asm volatile("s_wait_dscnt 0" ::: "memory");

        // ---- reload P as A-fragment (16x32 bf16) ----
        BF16Frag pf;
#pragma unroll
        for (int vv = 0; vv < 8; ++vv) {
            const int k0 = 2 * vv + 16 * hh;
            pf.u[vv] = *(const unsigned int*)&pstage[wave][lm][k0];
        }

        // ---- O += P * V  (8 dim-chunks) and l += P * ones ----
#pragma unroll
        for (int c = 0; c < 8; ++c) {
            BF16Frag vf;
#pragma unroll
            for (int vv = 0; vv < 8; ++vv) {
                const int kp = vv + 8 * hh;
                vf.u[vv] = vlds[buf][c * 16 + lm][kp];
            }
            o_acc[c] = __builtin_amdgcn_wmma_f32_16x16x32_bf16(false, pf.v, false, vf.v,
                                                               (short)0, o_acc[c], false, false);
        }
        o_sum = __builtin_amdgcn_wmma_f32_16x16x32_bf16(false, pf.v, false, ones_b.v,
                                                        (short)0, o_sum, false, false);
    }

    // ---- epilogue: O / l, coalesced 16-float runs per half-wave ----
    float* ob = outg + ((size_t)b * S_LEN + qBase) * D_DIM;
#pragma unroll
    for (int j = 0; j < 8; ++j) {
        const float ls  = __shfl(o_sum[j], hh * 16, 32);   // col 0 of the group
        const float inv = 1.0f / ls;
        const int qr = j + 8 * hh;
#pragma unroll
        for (int c = 0; c < 8; ++c) {
            ob[(size_t)qr * D_DIM + c * 16 + lm] = o_acc[c][j] * inv;
        }
    }
}

extern "C" void kernel_launch(void* const* d_in, const int* in_sizes, int n_in,
                              void* d_out, int out_size, void* d_ws, size_t ws_size,
                              hipStream_t stream) {
    (void)in_sizes; (void)n_in; (void)out_size; (void)d_ws; (void)ws_size;
    const float*         q    = (const float*)d_in[0];
    const float*         k    = (const float*)d_in[1];
    const float*         v    = (const float*)d_in[2];
    const unsigned char* mask = (const unsigned char*)d_in[3];
    float*               out  = (float*)d_out;

    dim3 grid(16 * (S_LEN / (QTILE * WAVES)));   // B * q-blocks = 256
    dim3 block(256);                              // 8 wave32 per block
    fattn_wmma_bf16<<<grid, block, 0, stream>>>(q, k, v, mask, out);
}
